// HybridTransformer_65481071395789
// MI455X (gfx1250) — matrix operand order repaired and checked by reading, not compiled, not run on hardware
//
#include <hip/hip_runtime.h>
#include <hip/hip_bf16.h>

// ---------------------------------------------------------------------------
// Model constants (match reference)
// ---------------------------------------------------------------------------
#define VOCAB 32000
#define DMODEL 512
#define NHEAD 8
#define HD 64
#define NLAYER 6
#define FFN 2048
#define NCLS 4
#define BATCH 16
#define SEQ 1024
#define NTOK (BATCH * SEQ)   // 16384
#define LN_EPS 1e-5f

typedef __attribute__((ext_vector_type(16))) __bf16 v16bf;
typedef __attribute__((ext_vector_type(8)))  __bf16 v8bf;
typedef __attribute__((ext_vector_type(8)))  float  v8f;

static __device__ inline v8f wmma_bf16(v16bf a, v16bf b, v8f c) {
    // D(16x16,f32) = A(16x32,bf16) x B(32x16,bf16) + C
    return __builtin_amdgcn_wmma_f32_16x16x32_bf16(
        false, a, false, b, (short)0, c, false, false);
}

static __device__ inline v16bf ld_frag2(const __bf16* p0, const __bf16* p1) {
    v16bf v;
    *((v8bf*)&v)     = *(const v8bf*)p0;
    *((v8bf*)&v + 1) = *(const v8bf*)p1;
    return v;
}

// ---------------------------------------------------------------------------
// Weight convert + repack into WMMA B-fragment-native layout.
//   Wp[l][kt][nt][lane][e] = bf16( W[l][kt*32 + (e>>3)*16 + (lane>>4)*8 + (e&7)][nt*16 + (lane&15)] )
// so each lane's 16 B-fragment elements are one contiguous 32-byte chunk.
// ---------------------------------------------------------------------------
__global__ __launch_bounds__(256) void k_pack_w(const float* __restrict__ src,
                                                __bf16* __restrict__ dst,
                                                int Lc, int K, int N) {
    const long per = (long)K * N;
    const long total = (long)Lc * per;
    const int ntiles = N >> 4;
    long o = (long)blockIdx.x * blockDim.x + threadIdx.x;
    const long stride = (long)gridDim.x * blockDim.x;
    for (; o < total; o += stride) {
        long r = o % per;
        long l = o / per;
        int e    = (int)(r & 15);
        int lane = (int)((r >> 4) & 31);
        long blk = r >> 9;                 // 32x16 k-n tile index
        int kt = (int)(blk / ntiles);
        int nt = (int)(blk - (long)kt * ntiles);
        int k = kt * 32 + ((e >> 3) << 4) + ((lane >> 4) << 3) + (e & 7);
        int n = (nt << 4) + (lane & 15);
        dst[o] = (__bf16)src[l * per + (long)k * N + n];
    }
}

// ---------------------------------------------------------------------------
// Embedding + sinusoidal positional encoding. One block per token.
// ---------------------------------------------------------------------------
__global__ __launch_bounds__(256) void k_embed(const int* __restrict__ x,
                                               const float* __restrict__ emb,
                                               float* __restrict__ h,
                                               __bf16* __restrict__ hb) {
    const long tok = blockIdx.x;
    const int  s   = (int)(tok % SEQ);
    const int  v   = x[tok];
    for (int d = threadIdx.x; d < DMODEL; d += blockDim.x) {
        float div = __expf((float)(d & ~1) * -0.0179889460f);   // ln(10000)/512
        float ang = (float)s * div;
        float pe  = (d & 1) ? __cosf(ang) : __sinf(ang);
        float val = emb[(long)v * DMODEL + d] + pe;
        h [tok * DMODEL + d] = val;
        hb[tok * DMODEL + d] = (__bf16)val;
    }
}

// ---------------------------------------------------------------------------
// bf16 WMMA GEMM, out = act(A[M,K] @ W[K,N] + bias).
// A row-major bf16; Wp fragment-packed (see k_pack_w).
// block = 256 = 8 waves; block tile 32 x 256; wave tile 32x32 (2x2 WMMA tiles).
// grid (M/32, N/256). Per K=32 step: 8x b128 loads -> 4 independent WMMAs.
// ---------------------------------------------------------------------------
__global__ __launch_bounds__(256) void k_gemm(const __bf16* __restrict__ A,
                                              const __bf16* __restrict__ Wp,
                                              const float*  __restrict__ bias,
                                              float*  __restrict__ outF,
                                              __bf16* __restrict__ outB,
                                              int M, int N, int K, int relu) {
    const int wave   = threadIdx.x >> 5;
    const int lane   = threadIdx.x & 31;
    const int laneLo = lane & 15;
    const int laneHi = lane >> 4;
    const int m0 = blockIdx.x * 32;
    const int n0 = blockIdx.y * 256 + wave * 16;      // 2nd n-tile at n0+128

    // A fragments: rows m0+laneLo and m0+16+laneLo; two 8-elem runs per row.
    const __bf16* arow0 = A + (long)(m0 + laneLo) * K + (laneHi << 3);
    const __bf16* arow1 = arow0 + (long)16 * K;
    // B fragments: packed; one 16-elem chunk per lane per (kt, nt).
    const long wstep = (long)(N >> 4) * 512;          // elems per k-tile row
    const __bf16* wl0 = Wp + ((long)(n0 >> 4) * 32 + lane) * 16;
    const __bf16* wl1 = Wp + ((long)((n0 + 128) >> 4) * 32 + lane) * 16;

    v8f acc00 = {}, acc01 = {}, acc10 = {}, acc11 = {};
    for (int k0 = 0; k0 < K; k0 += 32) {
        const long ws = (long)(k0 >> 5) * wstep;
        v16bf a0 = ld_frag2(arow0 + k0, arow0 + k0 + 16);
        v16bf a1 = ld_frag2(arow1 + k0, arow1 + k0 + 16);
        v16bf b0 = ld_frag2(wl0 + ws,   wl0 + ws + 8);
        v16bf b1 = ld_frag2(wl1 + ws,   wl1 + ws + 8);
        if (k0 + 32 < K) {
            __builtin_prefetch(arow0 + k0 + 32, 0, 3);
            __builtin_prefetch(wl0 + ws + wstep, 0, 3);
            __builtin_prefetch(wl1 + ws + wstep, 0, 3);
        }
        acc00 = wmma_bf16(a0, b0, acc00);
        acc01 = wmma_bf16(a0, b1, acc01);
        acc10 = wmma_bf16(a1, b0, acc10);
        acc11 = wmma_bf16(a1, b1, acc11);
    }

#pragma unroll
    for (int i = 0; i < 2; ++i) {
#pragma unroll
        for (int j = 0; j < 2; ++j) {
            const v8f acc = (i == 0) ? (j == 0 ? acc00 : acc01)
                                     : (j == 0 ? acc10 : acc11);
#pragma unroll
            for (int r = 0; r < 8; ++r) {
                int row = m0 + i * 16 + r + (laneHi << 3);
                int col = n0 + j * 128 + laneLo;
                float v = acc[r];
                if (bias) v += bias[col];
                if (relu) v = v > 0.f ? v : 0.f;
                long idx = (long)row * N + col;
                if (outF) outF[idx] = v;
                if (outB) outB[idx] = (__bf16)v;
            }
        }
    }
}

// ---------------------------------------------------------------------------
// Transpose V out of the qkv buffer: Vt[b][h][d][s] = qkv[b][s][2D + h*64 + d]
// makes the P@V B-fragment loads contiguous along the sequence dim.
// ---------------------------------------------------------------------------
__global__ __launch_bounds__(256) void k_transpose_v(const __bf16* __restrict__ qkv,
                                                     __bf16* __restrict__ vt) {
    const long total = (long)BATCH * NHEAD * HD * SEQ;
    long o = (long)blockIdx.x * blockDim.x + threadIdx.x;
    const long stride = (long)gridDim.x * blockDim.x;
    for (; o < total; o += stride) {
        int s = (int)(o & (SEQ - 1));
        int d = (int)((o >> 10) & (HD - 1));
        int h = (int)((o >> 16) & (NHEAD - 1));
        int b = (int)(o >> 19);
        vt[o] = qkv[((long)b * SEQ + s) * (3 * DMODEL) + 2 * DMODEL + h * HD + d];
    }
}

// ---------------------------------------------------------------------------
// Fused attention for one (batch, head, 16-query tile).
// grid = (SEQ/16, NHEAD, BATCH), block = 256 (8 waves).
// Dynamic LDS: f32 scores[16][1024] | bf16 probs[16][1024] | scratch.
// ---------------------------------------------------------------------------
__global__ __launch_bounds__(256) void k_attn(const __bf16* __restrict__ qkv,
                                              const __bf16* __restrict__ vt,
                                              __bf16* __restrict__ attnOut) {
    extern __shared__ char smem_raw[];
    float*  sc      = (float*)smem_raw;                         // 64 KB
    __bf16* pb      = (__bf16*)(smem_raw + 65536);              // 32 KB
    float*  red     = (float*)(smem_raw + 65536 + 32768);       // 1 KB
    float*  rowStat = red + 256;

    const int q0   = blockIdx.x * 16;
    const int hh   = blockIdx.y;
    const int bb   = blockIdx.z;
    const int wave = threadIdx.x >> 5;
    const int lane = threadIdx.x & 31;
    const int laneLo = lane & 15;
    const int laneHi = lane >> 4;
    const long rowStride = 3 * DMODEL;
    const __bf16* base = qkv + (long)bb * SEQ * rowStride;

    // ---- Q fragments (16x64 = 2 K-chunks), b128 loads, kept in registers ----
    const __bf16* qrow = base + (long)(q0 + laneLo) * rowStride + hh * HD + (laneHi << 4);
    v16bf aq0 = ld_frag2(qrow,      qrow + 8);
    v16bf aq1 = ld_frag2(qrow + 32, qrow + 40);

    // ---- Phase 1: scores = scale * Q K^T  (each wave: 8 of 64 key tiles) ----
    for (int kt = wave; kt < SEQ / 16; kt += 8) {
        const int kt0 = kt * 16;
        const __bf16* krow = base + (long)(kt0 + laneLo) * rowStride
                             + DMODEL + hh * HD + (laneHi << 4);
        v16bf b0 = ld_frag2(krow,      krow + 8);
        v16bf b1 = ld_frag2(krow + 32, krow + 40);
        v8f acc = {};
        acc = wmma_bf16(aq0, b0, acc);
        acc = wmma_bf16(aq1, b1, acc);
#pragma unroll
        for (int r = 0; r < 8; ++r)
            sc[(r + (laneHi << 3)) * SEQ + kt0 + laneLo] = acc[r] * 0.125f;
    }
    __syncthreads();

    // ---- Phase 2: fp32 row softmax -> bf16 probs (16 threads per row) ----
    {
        const int row = threadIdx.x >> 4;
        const int sub = threadIdx.x & 15;
        float mx = -1e30f;
        for (int c = sub; c < SEQ; c += 16) mx = fmaxf(mx, sc[row * SEQ + c]);
        red[row * 16 + sub] = mx;
        __syncthreads();
        if (sub == 0) {
            float m2 = red[row * 16];
            for (int i = 1; i < 16; ++i) m2 = fmaxf(m2, red[row * 16 + i]);
            rowStat[row] = m2;
        }
        __syncthreads();
        const float rmax = rowStat[row];
        float sum = 0.f;
        for (int c = sub; c < SEQ; c += 16) {
            float e = __expf(sc[row * SEQ + c] - rmax);
            sc[row * SEQ + c] = e;
            sum += e;
        }
        red[row * 16 + sub] = sum;
        __syncthreads();
        if (sub == 0) {
            float s2 = 0.f;
            for (int i = 0; i < 16; ++i) s2 += red[row * 16 + i];
            rowStat[row] = 1.0f / s2;
        }
        __syncthreads();
        const float rinv = rowStat[row];
        for (int c = sub; c < SEQ; c += 16)
            pb[row * SEQ + c] = (__bf16)(sc[row * SEQ + c] * rinv);
    }
    __syncthreads();

    // ---- Phase 3: attn(16x64) = P(16x1024) @ V(1024x64) ----
    // All 8 waves: wave&3 selects d-tile, wave>>2 selects K half (split-K);
    // two interleaved accumulators per wave shorten the WMMA chain to 8.
    {
        const int d0    = (wave & 3) * 16;
        const int kbase = (wave >> 2) * (SEQ / 2);
        const __bf16* prow = pb + laneLo * SEQ + (laneHi << 4) + kbase;   // LDS
        const __bf16* vrow = vt + (((long)bb * NHEAD + hh) * HD + d0 + laneLo) * SEQ
                             + (laneHi << 4) + kbase;
        v8f acc0 = {}, acc1 = {};
        for (int kk = 0; kk < SEQ / 2; kk += 64) {
            v16bf a0 = ld_frag2(prow + kk,      prow + kk + 8);
            v16bf b0 = ld_frag2(vrow + kk,      vrow + kk + 8);
            v16bf a1 = ld_frag2(prow + kk + 32, prow + kk + 40);
            v16bf b1 = ld_frag2(vrow + kk + 32, vrow + kk + 40);
            acc0 = wmma_bf16(a0, b0, acc0);
            acc1 = wmma_bf16(a1, b1, acc1);
        }
        v8f acc = acc0 + acc1;

        // combine the two K halves: waves 4-7 publish partials via LDS (reuse sc)
        float* partial = sc;   // 4 tiles * 32 lanes * 8 = 4 KB
        if (wave >= 4) {
#pragma unroll
            for (int r = 0; r < 8; ++r)
                partial[(((wave & 3) * 32 + lane) << 3) + r] = acc[r];
        }
        __syncthreads();
        if (wave < 4) {
#pragma unroll
            for (int r = 0; r < 8; ++r) {
                float v = acc[r] + partial[(((wave & 3) * 32 + lane) << 3) + r];
                attnOut[((long)bb * SEQ + q0 + r + (laneHi << 3)) * DMODEL
                        + hh * HD + d0 + laneLo] = (__bf16)v;
            }
        }
    }
}

// ---------------------------------------------------------------------------
// h = LayerNorm(h + t) * g + b  (in place; also emits bf16 copy).
// one block of 256 threads per row of 512; float2 accesses.
// ---------------------------------------------------------------------------
__global__ __launch_bounds__(256) void k_add_ln(float* __restrict__ h,
                                                const float* __restrict__ t,
                                                const float* __restrict__ g,
                                                const float* __restrict__ b,
                                                __bf16* __restrict__ hb) {
    __shared__ float part[256];
    const long row = blockIdx.x;
    const int  tid = threadIdx.x;
    float2* hr = (float2*)(h + row * DMODEL);
    const float2* tr = (const float2*)(t + row * DMODEL);

    float2 hv = hr[tid], tv = tr[tid];
    float v0 = hv.x + tv.x;
    float v1 = hv.y + tv.y;

    part[tid] = v0 + v1;
    __syncthreads();
    for (int s = 128; s > 0; s >>= 1) {
        if (tid < s) part[tid] += part[tid + s];
        __syncthreads();
    }
    const float mu = part[0] * (1.0f / DMODEL);
    __syncthreads();

    float d0 = v0 - mu, d1 = v1 - mu;
    part[tid] = d0 * d0 + d1 * d1;
    __syncthreads();
    for (int s = 128; s > 0; s >>= 1) {
        if (tid < s) part[tid] += part[tid + s];
        __syncthreads();
    }
    const float rstd = rsqrtf(part[0] * (1.0f / DMODEL) + LN_EPS);

    float2 gv = ((const float2*)g)[tid];
    float2 bv = ((const float2*)b)[tid];
    float o0 = d0 * rstd * gv.x + bv.x;
    float o1 = d1 * rstd * gv.y + bv.y;
    hr[tid] = make_float2(o0, o1);
    __bf16* hbp = hb + row * DMODEL + tid * 2;
    hbp[0] = (__bf16)o0;
    hbp[1] = (__bf16)o1;
}

// ---------------------------------------------------------------------------
// Mean pool over sequence + 4-class linear head. grid = BATCH, block = 512.
// ---------------------------------------------------------------------------
__global__ __launch_bounds__(512) void k_pool_cls(const float* __restrict__ h,
                                                  const float* __restrict__ clsw,
                                                  const float* __restrict__ clsb,
                                                  float* __restrict__ out) {
    __shared__ float pooled[DMODEL];
    const int bb = blockIdx.x;
    const int d  = threadIdx.x;
    float s = 0.f;
    for (int t = 0; t < SEQ; ++t) s += h[((long)bb * SEQ + t) * DMODEL + d];
    pooled[d] = s * (1.0f / SEQ);
    __syncthreads();
    if (d < NCLS) {
        float acc = clsb[d];
        for (int k = 0; k < DMODEL; ++k) acc += pooled[k] * clsw[k * NCLS + d];
        out[bb * NCLS + d] = acc;
    }
}

// ---------------------------------------------------------------------------
// Host orchestration
// ---------------------------------------------------------------------------
extern "C" void kernel_launch(void* const* d_in, const int* in_sizes, int n_in,
                              void* d_out, int out_size, void* d_ws, size_t ws_size,
                              hipStream_t stream) {
    const int*   x     = (const int*)  d_in[0];
    const float* emb   = (const float*)d_in[1];
    const float* qkv_w = (const float*)d_in[2];
    const float* out_w = (const float*)d_in[3];
    const float* out_b = (const float*)d_in[4];
    const float* w1    = (const float*)d_in[5];
    const float* b1    = (const float*)d_in[6];
    const float* w2    = (const float*)d_in[7];
    const float* b2    = (const float*)d_in[8];
    const float* ln1_g = (const float*)d_in[9];
    const float* ln1_b = (const float*)d_in[10];
    const float* ln2_g = (const float*)d_in[11];
    const float* ln2_b = (const float*)d_in[12];
    const float* cls_w = (const float*)d_in[13];
    const float* cls_b = (const float*)d_in[14];

    char* ws = (char*)d_ws;
    size_t off = 0;
    auto alloc = [&](size_t bytes) -> void* {
        void* p = ws + off;
        off = (off + bytes + 255) & ~(size_t)255;
        return p;
    };
    const long nQKVW = (long)NLAYER * DMODEL * 3 * DMODEL;
    const long nOUTW = (long)NLAYER * DMODEL * DMODEL;
    const long nW1   = (long)NLAYER * DMODEL * FFN;
    const long nW2   = (long)NLAYER * FFN * DMODEL;

    __bf16* qkvw_p  = (__bf16*)alloc(nQKVW * 2);
    __bf16* outw_p  = (__bf16*)alloc(nOUTW * 2);
    __bf16* w1_p    = (__bf16*)alloc(nW1 * 2);
    __bf16* w2_p    = (__bf16*)alloc(nW2 * 2);
    float*  h_f32   = (float*) alloc((long)NTOK * DMODEL * 4);
    __bf16* h_bf    = (__bf16*)alloc((long)NTOK * DMODEL * 2);
    __bf16* qkv_bf  = (__bf16*)alloc((long)NTOK * 3 * DMODEL * 2);
    __bf16* vt_bf   = (__bf16*)alloc((long)NTOK * DMODEL * 2);
    __bf16* attn_bf = (__bf16*)alloc((long)NTOK * DMODEL * 2);
    float*  tmp_f32 = (float*) alloc((long)NTOK * DMODEL * 4);
    __bf16* ffn1_bf = (__bf16*)alloc((long)NTOK * FFN * 2);
    (void)ws_size; (void)in_sizes; (void)n_in; (void)out_size;

    // ---- convert + fragment-pack all weights (once per call) ----
    k_pack_w<<<4096, 256, 0, stream>>>(qkv_w, qkvw_p, NLAYER, DMODEL, 3 * DMODEL);
    k_pack_w<<<4096, 256, 0, stream>>>(out_w, outw_p, NLAYER, DMODEL, DMODEL);
    k_pack_w<<<4096, 256, 0, stream>>>(w1,    w1_p,   NLAYER, DMODEL, FFN);
    k_pack_w<<<4096, 256, 0, stream>>>(w2,    w2_p,   NLAYER, FFN, DMODEL);

    // ---- embedding + PE ----
    k_embed<<<NTOK, 256, 0, stream>>>(x, emb, h_f32, h_bf);

    const size_t attn_lds = 65536 + 32768 + 1024 + 64;

    for (int l = 0; l < NLAYER; ++l) {
        // QKV projection: [16384,512] @ [512,1536] -> bf16
        k_gemm<<<dim3(NTOK / 32, (3 * DMODEL) / 256), 256, 0, stream>>>(
            h_bf, qkvw_p + (long)l * DMODEL * 3 * DMODEL, nullptr,
            nullptr, qkv_bf, NTOK, 3 * DMODEL, DMODEL, 0);

        // V transpose for contiguous P@V fragments
        k_transpose_v<<<4096, 256, 0, stream>>>(qkv_bf, vt_bf);

        // fused attention
        k_attn<<<dim3(SEQ / 16, NHEAD, BATCH), 256, attn_lds, stream>>>(
            qkv_bf, vt_bf, attn_bf);

        // output projection (+bias) -> fp32 tmp
        k_gemm<<<dim3(NTOK / 32, DMODEL / 256), 256, 0, stream>>>(
            attn_bf, outw_p + (long)l * DMODEL * DMODEL, out_b + (long)l * DMODEL,
            tmp_f32, nullptr, NTOK, DMODEL, DMODEL, 0);

        // residual + LN1
        k_add_ln<<<NTOK, 256, 0, stream>>>(h_f32, tmp_f32,
                                           ln1_g + (long)l * DMODEL,
                                           ln1_b + (long)l * DMODEL, h_bf);

        // FFN up (+bias, ReLU) -> bf16
        k_gemm<<<dim3(NTOK / 32, FFN / 256), 256, 0, stream>>>(
            h_bf, w1_p + (long)l * DMODEL * FFN, b1 + (long)l * FFN,
            nullptr, ffn1_bf, NTOK, FFN, DMODEL, 1);

        // FFN down (+bias) -> fp32 tmp
        k_gemm<<<dim3(NTOK / 32, DMODEL / 256), 256, 0, stream>>>(
            ffn1_bf, w2_p + (long)l * FFN * DMODEL, b2 + (long)l * DMODEL,
            tmp_f32, nullptr, NTOK, DMODEL, FFN, 0);

        // residual + LN2
        k_add_ln<<<NTOK, 256, 0, stream>>>(h_f32, tmp_f32,
                                           ln2_g + (long)l * DMODEL,
                                           ln2_b + (long)l * DMODEL, h_bf);
    }

    // mean pool + classifier head
    k_pool_cls<<<BATCH, 512, 0, stream>>>(h_f32, cls_w, cls_b, (float*)d_out);
}
